// GraphSAGE_65240553226754
// MI455X (gfx1250) — matrix operand order repaired and checked by reading, not compiled, not run on hardware
//
#include <hip/hip_runtime.h>
#include <hip/hip_bf16.h>

// ---------------------------------------------------------------------------
// GraphSAGE forward for MI455X (gfx1250, wave32, WMMA).
//   layer: h_out = relu( ((segsum(h[src]->dst) + h) / (deg+1)) @ W + b ) * mask
//   then max-pool over nodes per graph, then FC.
// GEMMs run on v_wmma_f32_16x16x32_bf16 (bf16 in / f32 accumulate).
// ---------------------------------------------------------------------------

typedef __attribute__((ext_vector_type(16))) __bf16 v16bf;
typedef __attribute__((ext_vector_type(8)))  float  v8f;

#define BB     8
#define NN     4096
#define EPER   16
#define BN     (BB * NN)          // 32768 nodes
#define NEDGE  (BB * NN * EPER)   // 524288 edges
#define INF0   128
#define H1F    64
#define H2F    32
#define OUTD   10

// ----------------------------- degree count -------------------------------
__global__ void deg_kernel(const int* __restrict__ dst, float* __restrict__ deg) {
    int t = blockIdx.x * blockDim.x + threadIdx.x;
    if (t < NEDGE) atomicAdd(&deg[dst[t]], 1.0f);
}

// ------------------------ edge scatter (segment_sum) -----------------------
// One thread = one edge x one 4-float feature chunk. float4 gather + 4 atomics.
template<int F>
__global__ void agg_kernel(const float* __restrict__ hin, const int* __restrict__ src,
                           const int* __restrict__ dst, float* __restrict__ agg) {
    const int chunks = F / 4;
    int t = blockIdx.x * blockDim.x + threadIdx.x;
    int e = t / chunks;
    int c = t - e * chunks;
    if (e >= NEDGE) return;
    int s = src[e], d = dst[e];
    const float4 v = *(const float4*)(hin + (size_t)s * F + c * 4);
    float* o = agg + (size_t)d * F + c * 4;
    atomicAdd(o + 0, v.x);
    atomicAdd(o + 1, v.y);
    atomicAdd(o + 2, v.z);
    atomicAdd(o + 3, v.w);
}

// ---------------- fused normalize + GEMM(WMMA) + bias/relu/mask ------------
// Workgroup = 16 output rows, NF/16 waves; wave w computes N-tile [16w,16w+16).
// A (16xKF) and W^T (NFxKF) staged in LDS as bf16; fragments assembled per the
// CDNA5 ISA 16-bit WMMA VGPR layouts (ISA 7.12.2).
template<int KF, int NF>
__global__ __launch_bounds__(NF * 2)
void sage_layer_wmma(const float* __restrict__ hin,  // [BN,KF]
                     const float* __restrict__ agg,  // [BN,KF]
                     const float* __restrict__ deg,  // [BN]
                     const float* __restrict__ mask, // [BN]
                     const float* __restrict__ W,    // [KF,NF]
                     const float* __restrict__ bias, // [NF]
                     float* __restrict__ hout)       // [BN,NF]
{
    __shared__ __bf16 sA[16][KF];    // normalized node features, bf16
    __shared__ __bf16 sWt[NF][KF];   // W transposed, bf16 (pair K reads contiguous)
    __shared__ float  sMask[16];

    const int tid  = threadIdx.x;
    const int nthr = NF * 2;         // (NF/16) waves * 32 lanes
    const int row0 = blockIdx.x * 16;

    // Stage A tile: (agg + h) / (deg + 1), converted to bf16.
    for (int i = tid; i < 16 * KF; i += nthr) {
        int m = i / KF, k = i - m * KF;
        int r = row0 + m;
        float v = (agg[(size_t)r * KF + k] + hin[(size_t)r * KF + k]) / (deg[r] + 1.0f);
        sA[m][k] = (__bf16)v;
    }
    // Stage W^T.
    for (int i = tid; i < KF * NF; i += nthr) {
        int k = i / NF, n = i - k * NF;
        sWt[n][k] = (__bf16)W[(size_t)k * NF + n];
    }
    if (tid < 16) sMask[tid] = mask[row0 + tid];
    __syncthreads();

    const int wave = tid >> 5;
    const int lane = tid & 31;
    const int m    = lane & 15;      // A row / B,C,D column index within tile
    const int g    = lane >> 4;      // half-wave group
    const int ncol = wave * 16 + m;  // output column

    v8f c = {0.f, 0.f, 0.f, 0.f, 0.f, 0.f, 0.f, 0.f};
    #pragma unroll
    for (int kb = 0; kb < KF; kb += 32) {
        v16bf afrag, bfrag;
        // A 16x32 bf16: halves 0..7 -> K = kb+8g+h ; halves 8..15 -> K = kb+16+8g+h
        #pragma unroll
        for (int h = 0; h < 8; ++h) {
            afrag[h]     = sA[m][kb + 8 * g + h];
            afrag[h + 8] = sA[m][kb + 16 + 8 * g + h];
        }
        // B 32x16 bf16: VGPR p holds K = kb+8g+2p+(p>=4?8:0) (lo), +1 (hi); lane n = ncol
        #pragma unroll
        for (int p = 0; p < 8; ++p) {
            bfrag[2 * p]     = sWt[ncol][kb + 8 * g + 2 * p + (p >= 4 ? 8 : 0)];
            bfrag[2 * p + 1] = sWt[ncol][kb + 8 * g + 2 * p + 1 + (p >= 4 ? 8 : 0)];
        }
        c = __builtin_amdgcn_wmma_f32_16x16x32_bf16(
                /*neg_a=*/false, afrag, /*neg_b=*/false, bfrag,
                /*c_mod=*/(short)0, c, /*reuse_a=*/false, /*reuse_b=*/false);
    }

    // Epilogue: bias + relu + mask. C/D layout: row = p + 8g, col = ncol.
    const float bn = bias[ncol];
    #pragma unroll
    for (int p = 0; p < 8; ++p) {
        int mr = p + 8 * g;
        float val = fmaxf(c[p] + bn, 0.0f) * sMask[mr];
        hout[(size_t)(row0 + mr) * NF + ncol] = val;
    }
}

// ----------------------------- max pool per graph --------------------------
__global__ void maxpool_kernel(const float* __restrict__ h2, float* __restrict__ pooled) {
    __shared__ float red[256];
    int b = blockIdx.x;
    int f = threadIdx.x & 31;
    int g = threadIdx.x >> 5;       // 8 row-groups
    float m = 0.0f;                 // relu*mask output is >= 0
    for (int r = g; r < NN; r += 8)
        m = fmaxf(m, h2[((size_t)b * NN + r) * H2F + f]);
    red[threadIdx.x] = m;
    __syncthreads();
    if (g == 0) {
        float mm = red[f];
        #pragma unroll
        for (int gg = 1; gg < 8; ++gg) mm = fmaxf(mm, red[gg * 32 + f]);
        pooled[b * H2F + f] = mm;
    }
}

// ----------------------------- final FC (8x32 @ 32x10) ---------------------
__global__ void fc_kernel(const float* __restrict__ pooled, const float* __restrict__ Wfc,
                          const float* __restrict__ bfc, float* __restrict__ out) {
    int t = threadIdx.x;
    if (t >= BB * OUTD) return;
    int bb = t / OUTD, o = t - bb * OUTD;
    float acc = bfc[o];
    #pragma unroll
    for (int k = 0; k < H2F; ++k) acc += pooled[bb * H2F + k] * Wfc[k * OUTD + o];
    out[t] = acc;
}

// ---------------------------------------------------------------------------
extern "C" void kernel_launch(void* const* d_in, const int* in_sizes, int n_in,
                              void* d_out, int out_size, void* d_ws, size_t ws_size,
                              hipStream_t stream) {
    const float* x    = (const float*)d_in[0];
    const float* mask = (const float*)d_in[1];
    const int*   src  = (const int*)d_in[2];
    const int*   dst  = (const int*)d_in[3];
    const float* W1   = (const float*)d_in[4];
    const float* b1   = (const float*)d_in[5];
    const float* W2   = (const float*)d_in[6];
    const float* b2   = (const float*)d_in[7];
    const float* Wfc  = (const float*)d_in[8];
    const float* bfc  = (const float*)d_in[9];
    float* out = (float*)d_out;

    // Workspace layout (floats): deg | agg1 | h1 | agg2 | h2 | pooled  (~38 MB)
    float* ws     = (float*)d_ws;
    float* deg    = ws;
    float* agg1   = deg  + BN;
    float* h1     = agg1 + (size_t)BN * INF0;
    float* agg2   = h1   + (size_t)BN * H1F;
    float* h2     = agg2 + (size_t)BN * H1F;
    float* pooled = h2   + (size_t)BN * H2F;

    // Zero the accumulation buffers (deg+agg1 contiguous, agg2 separate).
    hipMemsetAsync(deg,  0, ((size_t)BN + (size_t)BN * INF0) * sizeof(float), stream);
    hipMemsetAsync(agg2, 0, (size_t)BN * H1F * sizeof(float), stream);

    deg_kernel<<<(NEDGE + 255) / 256, 256, 0, stream>>>(dst, deg);

    {
        int total = NEDGE * (INF0 / 4);
        agg_kernel<INF0><<<(total + 255) / 256, 256, 0, stream>>>(x, src, dst, agg1);
    }
    sage_layer_wmma<INF0, H1F><<<BN / 16, H1F * 2, 0, stream>>>(x, agg1, deg, mask, W1, b1, h1);

    {
        int total = NEDGE * (H1F / 4);
        agg_kernel<H1F><<<(total + 255) / 256, 256, 0, stream>>>(h1, src, dst, agg2);
    }
    sage_layer_wmma<H1F, H2F><<<BN / 16, H2F * 2, 0, stream>>>(h1, agg2, deg, mask, W2, b2, h2);

    maxpool_kernel<<<BB, 256, 0, stream>>>(h2, pooled);
    fc_kernel<<<1, 128, 0, stream>>>(pooled, Wfc, bfc, out);
}
